// FCOSLoss_80118319939964
// MI455X (gfx1250) — compile-verified, matrix operand order checked by reading, not run on hardware
//
#include <hip/hip_runtime.h>

// Problem dims (match reference)
constexpr int Bn = 4, Nn = 16, Hn = 512, Wn = 512;
constexpr int HW = Hn * Wn;
constexpr int MAXE = 32768;          // max crop elements, pow2, fits 320KB LDS

// ws layout (floats): [0,192) stats S/SS/C per (b,n); [256,1280) params; [1536,1600) lovasz
constexpr int WS_STATS  = 0;
constexpr int WS_PARAMS = 256;
constexpr int WS_LOV    = 1536;
constexpr int WS_ZERO_N = 2048;

typedef __attribute__((ext_vector_type(16))) __bf16 v16bf;
typedef __attribute__((ext_vector_type(8)))  float  v8f;

__device__ __forceinline__ float gridv(int i) { return 2.0f * (float)i / 2047.0f; }

// order-preserving float->uint (ascending uint == ascending float)
__device__ __forceinline__ unsigned fkey(float f) {
    unsigned u = __float_as_uint(f);
    return (u & 0x80000000u) ? ~u : (u | 0x80000000u);
}
__device__ __forceinline__ float unkey(unsigned k) {
    return (k & 0x80000000u) ? __uint_as_float(k ^ 0x80000000u) : __uint_as_float(~k);
}

__global__ void ae_zero(float* ws) {
    int i = blockIdx.x * blockDim.x + threadIdx.x;
    if (i < WS_ZERO_N) ws[i] = 0.0f;
}

// ---------------------------------------------------------------------------
// Stage 1: masked reductions via WMMA bf16, fed by the async global->LDS
// engine (double-buffered 1024-pixel tiles, paced with s_wait_asynccnt).
// Per wave: A = mask bits (16 instances x 32 pixels), B = (32 pixels x 16 cols)
// with col0 = sigma, col1 = sigma^2, col2 = 1.  D accumulates S / SS / C.
// ---------------------------------------------------------------------------
__global__ __launch_bounds__(32) void ae_stats_wmma(const float* __restrict__ ae,
                                                    const int*   __restrict__ inst,
                                                    float* ws) {
    constexpr int SUB  = 1024;                 // pixels per LDS tile
    constexpr int NSUB = 4;                    // tiles per block
    constexpr int BLKS = HW / (SUB * NSUB);    // 64 blocks per image

    __shared__ int   sIm[2][SUB];
    __shared__ float sSg[2][SUB];

    int blk  = blockIdx.x;
    int b    = blk / BLKS;
    int c    = blk % BLKS;
    int lane = threadIdx.x;
    size_t base = (size_t)c * (SUB * NSUB);

    const float* sig = ae + (size_t)b * 3 * HW + 2 * (size_t)HW;  // sigma channel
    const int*   im  = inst + (size_t)b * HW;

    // Issue one tile: 8 x b128 per lane per array -> 16 async ops total
    auto issue = [&](int buf, int sc) {
        size_t p0 = base + (size_t)sc * SUB;
        #pragma unroll
        for (int i = 0; i < 8; ++i) {
            int idx = lane * 4 + i * 128;
            unsigned dI = (unsigned)(uintptr_t)&sIm[buf][idx];
            unsigned dS = (unsigned)(uintptr_t)&sSg[buf][idx];
            unsigned long long gI = (unsigned long long)(uintptr_t)(im  + p0 + idx);
            unsigned long long gS = (unsigned long long)(uintptr_t)(sig + p0 + idx);
            asm volatile("global_load_async_to_lds_b128 %0, %1, off"
                         :: "v"(dI), "v"(gI) : "memory");
            asm volatile("global_load_async_to_lds_b128 %0, %1, off"
                         :: "v"(dS), "v"(gS) : "memory");
        }
    };

    v8f acc0 = {0.f, 0.f, 0.f, 0.f, 0.f, 0.f, 0.f, 0.f};
    v8f acc1 = {0.f, 0.f, 0.f, 0.f, 0.f, 0.f, 0.f, 0.f};
    int lo = lane & 15;        // A row (instance) == B column (stat)
    int half = lane >> 4;
    // branch-free B-column weights: col0 = sigma, col1 = sigma^2, col2 = 1
    float w0 = (lo == 0) ? 1.f : 0.f;
    float w1 = (lo == 1) ? 1.f : 0.f;
    float w2 = (lo == 2) ? 1.f : 0.f;

    issue(0, 0);
    for (int sc = 0; sc < NSUB; ++sc) {
        int cur = sc & 1;
        if (sc + 1 < NSUB) {
            issue(cur ^ 1, sc + 1);                               // next tile in flight
            asm volatile("s_wait_asynccnt 0x10" ::: "memory");    // cur tile landed
        } else {
            asm volatile("s_wait_asynccnt 0x0" ::: "memory");
        }
        __builtin_prefetch(sig + base + (size_t)(sc + 2) * SUB, 0, 1);  // warm L2

        for (int g = 0; g < SUB; g += 64) {
            v16bf a0, b0, a1, b1;
            #pragma unroll
            for (int t = 0; t < 16; ++t) {
                // A 16x32 bf16 layout: lane m: K=0..7,16..23 ; lane m+16: K=8..15,24..31
                int kk = ((t < 8) ? t : t + 8) + half * 8;
                int k0 = g + kk, k1 = g + 32 + kk;
                a0[t] = (__bf16)((sIm[cur][k0] == lo + 1) ? 1.0f : 0.0f);
                a1[t] = (__bf16)((sIm[cur][k1] == lo + 1) ? 1.0f : 0.0f);
                float s0 = sSg[cur][k0], s1 = sSg[cur][k1];
                b0[t] = (__bf16)fmaf(w1, s0 * s0, fmaf(w0, s0, w2));
                b1[t] = (__bf16)fmaf(w1, s1 * s1, fmaf(w0, s1, w2));
            }
            acc0 = __builtin_amdgcn_wmma_f32_16x16x32_bf16(
                       false, a0, false, b0, (short)0, acc0, false, false);
            acc1 = __builtin_amdgcn_wmma_f32_16x16x32_bf16(
                       false, a1, false, b1, (short)0, acc1, false, false);
        }
    }

    // D 16x16 f32: lane holds column N = lane&15; VGPR r holds row r + 8*(lane>=16)
    int ncol = lane & 15;
    if (ncol < 3) {
        #pragma unroll
        for (int r = 0; r < 8; ++r) {
            int m = r + 8 * half;
            atomicAdd(&ws[WS_STATS + b * 48 + ncol * 16 + m], acc0[r] + acc1[r]);
        }
    }
}

// ---------------------------------------------------------------------------
// Stage 2: per-(b,n) scalars: var_loss, s_exp, validity, box geometry.
// ---------------------------------------------------------------------------
__global__ void ae_prep(const int* __restrict__ boxes, float* ws) {
    int t = threadIdx.x;
    if (t >= Bn * Nn) return;
    int b = t / Nn, n = t % Nn;
    float S  = ws[WS_STATS + b * 48 + n];
    float SS = ws[WS_STATS + b * 48 + 16 + n];
    float C  = ws[WS_STATS + b * 48 + 32 + n];
    float cnt = fmaxf(C, 1.0f);
    float sd  = S / cnt;                                    // s_mean (stop-grad value)
    float var = (SS - 2.0f * sd * S + sd * sd * C) / cnt;   // Σ m(σ-sd)² / cnt
    float sexp  = expf(sd);
    float valid = (C > 0.5f) ? 1.0f : 0.0f;

    const int* bx = boxes + (size_t)t * 4;
    float y1 = (float)bx[0], x1 = (float)bx[1], y2 = (float)bx[2], x2 = (float)bx[3];
    int cy = (int)((y1 + y2) * 0.5f);
    int cx = (int)((x1 + x2) * 0.5f);
    float cyg = gridv(cy), cxg = gridv(cx);
    float cyf = (y1 + y2) * 0.5f, cxf = (x1 + x2) * 0.5f;
    float hy = (y2 - y1) * 0.5f * 1.5f, hx = (x2 - x1) * 0.5f * 1.5f;
    float ltY = fminf(fmaxf(floorf(cyf - hy), 0.f), (float)Hn);
    float rbY = fminf(fmaxf(ceilf (cyf + hy), 0.f), (float)Hn);
    float ltX = fminf(fmaxf(floorf(cxf - hx), 0.f), (float)Wn);
    float rbX = fminf(fmaxf(ceilf (cxf + hx), 0.f), (float)Wn);

    float* prm = ws + WS_PARAMS + t * 16;
    prm[0] = sexp; prm[1] = var; prm[2] = valid;
    prm[3] = cxg;  prm[4] = cyg;
    prm[5] = ltY;  prm[6] = rbY; prm[7] = ltX; prm[8] = rbX;
}

// ---------------------------------------------------------------------------
// Stage 3: Lovász hinge per (b,n).  Only crop pixels have errors > 0, so we
// sort just the crop (≤ ~18K elems) fully inside LDS: 32K-key bitonic sort,
// then segmented label prefix-sum + closed-form Jaccard gradient dot.
// ---------------------------------------------------------------------------
__global__ __launch_bounds__(1024) void ae_lovasz(const float* __restrict__ ae,
                                                  const int*   __restrict__ inst,
                                                  float* ws) {
    __shared__ unsigned keys[MAXE];     // 128 KB of the 320 KB WGP LDS
    __shared__ unsigned scn[1024];
    __shared__ float    red[1024];

    int bn = blockIdx.x;
    int b = bn / Nn, n = bn % Nn;
    int tid = threadIdx.x;

    const float* prm = ws + WS_PARAMS + bn * 16;
    float sexp = prm[0], cxg = prm[3], cyg = prm[4];
    int ltY = (int)prm[5], rbY = (int)prm[6], ltX = (int)prm[7], rbX = (int)prm[8];
    int ch = rbY - ltY, cw = rbX - ltX;
    int count = (ch > 0 && cw > 0) ? ch * cw : 0;
    if (count > MAXE) count = MAXE;
    if (count == 0) { if (tid == 0) ws[WS_LOV + bn] = 0.0f; return; }

    unsigned Pn = 1024;
    while ((int)Pn < count) Pn <<= 1;

    const float* ae0 = ae + (size_t)b * 3 * HW;       // x-embedding channel
    const float* ae1 = ae0 + HW;                      // y-embedding channel
    const int*   im  = inst + (size_t)b * HW;

    // Build keys: error (bits 31..1, order-preserving) | label (bit 0); pad = 0 (= -inf)
    for (unsigned i = tid; i < Pn; i += 1024) {
        unsigned kk = 0;
        if ((int)i < count) {
            int y = ltY + (int)i / cw;
            int x = ltX + (int)i % cw;
            int p = y * Wn + x;
            float ex = tanhf(ae0[p]) + gridv(x);
            float ey = tanhf(ae1[p]) + gridv(y);
            float dx = ex - cxg, dy = ey - cyg;
            float d  = expf(-sexp * (dx * dx + dy * dy));
            unsigned lab = (im[p] == n + 1) ? 1u : 0u;
            float err = lab ? (2.0f - 2.0f * d) : (2.0f * d);   // 1 - logit*sign
            kk = (fkey(err) & ~1u) | lab;
        }
        keys[i] = kk;
    }
    __syncthreads();

    // In-LDS bitonic sort, descending
    for (unsigned k = 2; k <= Pn; k <<= 1) {
        for (unsigned j = k >> 1; j > 0; j >>= 1) {
            for (unsigned i = tid; i < Pn; i += 1024) {
                unsigned ixj = i ^ j;
                if (ixj > i) {
                    unsigned va = keys[i], vb = keys[ixj];
                    bool up = ((i & k) == 0);
                    bool sw = up ? (va < vb) : (va > vb);
                    if (sw) { keys[i] = vb; keys[ixj] = va; }
                }
            }
            __syncthreads();
        }
    }

    // Segmented inclusive label prefix sum (each thread owns Pn/1024 contiguous elems)
    unsigned seg = Pn / 1024;
    unsigned s0  = (unsigned)tid * seg;
    unsigned lc = 0;
    for (unsigned s = 0; s < seg; ++s) lc += keys[s0 + s] & 1u;
    scn[tid] = lc;
    __syncthreads();
    for (unsigned off = 1; off < 1024; off <<= 1) {
        unsigned v   = scn[tid];
        unsigned add = (tid >= (int)off) ? scn[tid - off] : 0u;
        __syncthreads();
        scn[tid] = v + add;
        __syncthreads();
    }
    float gts = (float)scn[1023];
    unsigned cum = scn[tid] - lc;        // exclusive base for this segment

    // dot(relu(errors_sorted), grad) with closed-form jac(i, cum): no serial dependency
    float dot = 0.0f;
    for (unsigned s = 0; s < seg; ++s) {
        unsigned i  = s0 + s;
        unsigned kk = keys[i];
        unsigned lab = kk & 1u;
        cum += lab;
        float e  = fmaxf(unkey(kk), 0.0f);
        float fc = (float)cum;
        float jac = 1.0f - (gts - fc) / (gts + (float)(i + 1) - fc);
        float jprev = 0.0f;
        if (i != 0) {
            float fcp = (float)(cum - lab);
            jprev = 1.0f - (gts - fcp) / (gts + (float)i - fcp);
        }
        dot += e * (jac - jprev);
    }
    red[tid] = dot;
    __syncthreads();
    for (int s = 512; s > 0; s >>= 1) {
        if (tid < s) red[tid] += red[tid + s];
        __syncthreads();
    }
    if (tid == 0) ws[WS_LOV + bn] = red[0];
}

// ---------------------------------------------------------------------------
// Stage 4: final scalar
// ---------------------------------------------------------------------------
__global__ void ae_final(const float* __restrict__ ws, float* out) {
    __shared__ float sv[64], sc[64];
    int t = threadIdx.x;
    if (t < Bn * Nn) {
        const float* prm = ws + WS_PARAMS + t * 16;
        float var = prm[1], valid = prm[2];
        float lov = ws[WS_LOV + t];
        sv[t] = (var + lov) * valid;
        sc[t] = valid;
    }
    __syncthreads();
    if (t < Bn) {
        float s = 0.f, c = 0.f;
        for (int j = 0; j < Nn; ++j) { s += sv[t * Nn + j]; c += sc[t * Nn + j]; }
        sv[t] = s / fmaxf(c, 1.0f);
    }
    __syncthreads();
    if (t == 0) out[0] = 0.25f * (sv[0] + sv[1] + sv[2] + sv[3]);
}

extern "C" void kernel_launch(void* const* d_in, const int* in_sizes, int n_in,
                              void* d_out, int out_size, void* d_ws, size_t ws_size,
                              hipStream_t stream) {
    const float* ae    = (const float*)d_in[0];   // (4,3,512,512) f32
    const int*   inst  = (const int*)  d_in[1];   // (4,512,512) i32
    const int*   boxes = (const int*)  d_in[2];   // (4,16,4) i32
    float* out = (float*)d_out;
    float* ws  = (float*)d_ws;

    ae_zero      <<<(WS_ZERO_N + 255) / 256, 256, 0, stream>>>(ws);
    ae_stats_wmma<<<Bn * 64, 32, 0, stream>>>(ae, inst, ws);
    ae_prep      <<<1, 64, 0, stream>>>(boxes, ws);
    ae_lovasz    <<<Bn * Nn, 1024, 0, stream>>>(ae, inst, ws);
    ae_final     <<<1, 64, 0, stream>>>(ws, out);
}
